// AttentionC_45578192945457
// MI455X (gfx1250) — matrix operand order repaired and checked by reading, not compiled, not run on hardware
//
#include <hip/hip_runtime.h>
#include <hip/hip_bf16.h>

// ---------------------------------------------------------------------------
// AttentionC block for MI455X (gfx1250): GroupNorm -> QKV 1x1conv ->
// channel-attention (softmax over channels) -> out 1x1conv -> +residual.
// All GEMMs run on v_wmma_f32_16x16x32_f16 (f16 operands, f32 accum).
// Wave tile = 1x4 (16M x 64N): keeps live VGPRs ~60 so the compiler can
// pipeline loads into the WMMA stream without spilling (round-2 lesson).
// ---------------------------------------------------------------------------

typedef __attribute__((ext_vector_type(16))) _Float16 v16h;
typedef __attribute__((ext_vector_type(8)))  _Float16 v8h;
typedef __attribute__((ext_vector_type(8)))  float    v8f;

#define NB   16    // batch
#define NC   768   // channels
#define NHW  1024  // H*W
#define NG   32    // groups
#define CPG  24    // channels per group

// ---- WMMA fragment loads (contiguous K order, same in A and B) --------

// A matrix, row-major [M x K], 16x32 f16 fragment at (m0, k0).
// lane L: m = L%16; half=L/16; elem e -> K=k0+e+16*half
static __device__ __forceinline__
v16h load_frag_a(const _Float16* __restrict__ base, int ld, int m0, int k0, int lane) {
  const int m    = m0 + (lane & 15);
  const int half = lane >> 4;
  const _Float16* p = base + (size_t)m * ld + (k0 + 16 * half);
  v8h lo = *(const v8h*)(p);        // K = k0+16h .. +7
  v8h hi = *(const v8h*)(p + 8);    // K = k0+16h+8 .. +7
  return __builtin_shufflevector(lo, hi, 0,1,2,3,4,5,6,7,8,9,10,11,12,13,14,15);
}

// B matrix stored N-major [N x K] (contiguous K per n), 32x16 f16 fragment.
// lane L: n = n0 + L%16; elem e -> K = k0 + e + 16*(L/16)  => one 32B load.
static __device__ __forceinline__
v16h load_frag_b(const _Float16* __restrict__ base, int ld, int n0, int k0, int lane) {
  const int n    = n0 + (lane & 15);
  const int half = lane >> 4;
  return *(const v16h*)(base + (size_t)n * ld + (k0 + 16 * half));
}

static __device__ __forceinline__ v8f wmma16(v16h a, v16h b, v8f c) {
  return __builtin_amdgcn_wmma_f32_16x16x32_f16(false, a, false, b, (short)0, c,
                                                false, false);
}

// ---- K1: weight f32 -> f16 -------------------------------------------------
__global__ void cvt_weights_kernel(const float* __restrict__ wq,
                                   const float* __restrict__ wo,
                                   _Float16* __restrict__ wq16,
                                   _Float16* __restrict__ wo16) {
  const int NQ = 3 * NC * NC;          // 1769472
  const int NO = NC * NC;              // 589824
  int idx = blockIdx.x * 256 + threadIdx.x;
  if (idx < NQ)            wq16[idx]      = (_Float16)wq[idx];
  else if (idx < NQ + NO)  wo16[idx - NQ] = (_Float16)wo[idx - NQ];
}

// ---- K2: GroupNorm, writes xn TRANSPOSED as f16 [b][p][c] -----------------
__global__ void groupnorm_kernel(const float* __restrict__ x,
                                 const float* __restrict__ gamma,
                                 const float* __restrict__ beta,
                                 _Float16* __restrict__ xnt) {
  const int g = blockIdx.x;            // 0..31
  const int b = blockIdx.y;            // 0..15
  const int tid = threadIdx.x;         // 0..255
  const float* __restrict__ xg = x + ((size_t)b * NC + (size_t)g * CPG) * NHW;

  float s = 0.f, s2 = 0.f;
  for (int c = 0; c < CPG; ++c)
    for (int p = tid; p < NHW; p += 256) {
      float v = xg[c * NHW + p];
      s += v; s2 += v * v;
    }
  __shared__ float rs[256], rs2[256];
  rs[tid] = s; rs2[tid] = s2;
  __syncthreads();
  for (int off = 128; off > 0; off >>= 1) {
    if (tid < off) { rs[tid] += rs[tid + off]; rs2[tid] += rs2[tid + off]; }
    __syncthreads();
  }
  __shared__ float s_mean, s_rstd;
  if (tid == 0) {
    const float inv_n = 1.0f / (float)(CPG * NHW);
    float mean = rs[0] * inv_n;
    float var  = rs2[0] * inv_n - mean * mean;
    s_mean = mean;
    s_rstd = rsqrtf(var + 1e-5f);
  }
  __syncthreads();
  const float mean = s_mean, rstd = s_rstd;

  for (int p = tid; p < NHW; p += 256) {
    v8h hv[3];
    #pragma unroll
    for (int c = 0; c < CPG; ++c) {
      const float gm = gamma[g * CPG + c];
      const float bt = beta[g * CPG + c];
      hv[c >> 3][c & 7] = (_Float16)((xg[c * NHW + p] - mean) * rstd * gm + bt);
    }
    _Float16* dst = xnt + ((size_t)b * NHW + p) * NC + g * CPG;  // 16B aligned
    ((v8h*)dst)[0] = hv[0];
    ((v8h*)dst)[1] = hv[1];
    ((v8h*)dst)[2] = hv[2];
  }
}

// ---- K3: QKV GEMM  qkv = w_qkv[3C,C] @ xn[C,HW] + b  ----------------------
// A = w_qkv f16 row-major (ld C). B = xnt [p][c] (ld C).
// q,k stored [b][c][hw]; v stored transposed [b][hw][c].
__global__ void qkv_gemm_kernel(const _Float16* __restrict__ wq,
                                const _Float16* __restrict__ xnt,
                                const float* __restrict__ bias,
                                _Float16* __restrict__ qo,
                                _Float16* __restrict__ ko,
                                _Float16* __restrict__ vt) {
  const int b    = blockIdx.z;
  const int lane = threadIdx.x & 31;
  const int wave = threadIdx.x >> 5;
  const int m0   = blockIdx.y * 32  + (wave >> 2) * 16;   // 0..2288
  const int n0   = blockIdx.x * 256 + (wave & 3) * 64;    // 0..960
  const _Float16* __restrict__ Bm = xnt + (size_t)b * NHW * NC;

  v8f acc[4] = {};
  for (int k0 = 0; k0 < NC; k0 += 32) {
    v16h af  = load_frag_a(wq, NC, m0, k0, lane);
    v16h bf0 = load_frag_b(Bm, NC, n0,      k0, lane);
    v16h bf1 = load_frag_b(Bm, NC, n0 + 16, k0, lane);
    v16h bf2 = load_frag_b(Bm, NC, n0 + 32, k0, lane);
    v16h bf3 = load_frag_b(Bm, NC, n0 + 48, k0, lane);
    if (k0 + 32 < NC)
      __builtin_prefetch(wq + (size_t)(m0 + (lane & 15)) * NC + k0 + 32, 0, 1);
    acc[0] = wmma16(af, bf0, acc[0]);
    acc[1] = wmma16(af, bf1, acc[1]);
    acc[2] = wmma16(af, bf2, acc[2]);
    acc[3] = wmma16(af, bf3, acc[3]);
  }

  const int half = lane >> 4, lc = lane & 15;
  const int region = m0 / NC;   // 0=q 1=k 2=v (tiles never straddle: 768%16==0)
  #pragma unroll
  for (int t = 0; t < 4; ++t) {
    const int n = n0 + 16 * t + lc;
    #pragma unroll
    for (int r = 0; r < 8; ++r) {
      const int m   = m0 + r + 8 * half;
      const float v = acc[t][r] + bias[m];
      if (region == 0)
        qo[((size_t)b * NC + m) * NHW + n] = (_Float16)v;
      else if (region == 1)
        ko[((size_t)b * NC + (m - NC)) * NHW + n] = (_Float16)v;
      else
        vt[((size_t)b * NHW + n) * NC + (m - 2 * NC)] = (_Float16)v;
    }
  }
}

// ---- K4: att = softmax(q @ k^T * hw^-0.5) ; fused per 16-row panel --------
// A = q[b][i][d] (ld HW, contiguous K=d). B = k[b][j][d] (ld HW, contiguous d).
__global__ void att_softmax_kernel(const _Float16* __restrict__ q,
                                   const _Float16* __restrict__ k,
                                   _Float16* __restrict__ att) {
  const int b    = blockIdx.y;
  const int m0   = blockIdx.x * 16;        // row panel
  const int lane = threadIdx.x & 31;
  const int wave = threadIdx.x >> 5;
  const _Float16* __restrict__ qb = q + (size_t)b * NC * NHW;
  const _Float16* __restrict__ kb = k + (size_t)b * NC * NHW;

  v8f acc[6] = {};
  for (int k0 = 0; k0 < NHW; k0 += 32) {
    v16h af = load_frag_a(qb, NHW, m0, k0, lane);
    v16h bf[6];
    #pragma unroll
    for (int t = 0; t < 6; ++t)
      bf[t] = load_frag_b(kb, NHW, (wave * 6 + t) * 16, k0, lane);
    #pragma unroll
    for (int t = 0; t < 6; ++t) acc[t] = wmma16(af, bf[t], acc[t]);
  }

  __shared__ float lgt[16][NC];            // 48 KB
  const float scale = 0.03125f;            // 1024^-0.5
  const int half = lane >> 4, lc = lane & 15;
  #pragma unroll
  for (int t = 0; t < 6; ++t) {
    const int n = (wave * 6 + t) * 16 + lc;
    #pragma unroll
    for (int r = 0; r < 8; ++r)
      lgt[r + 8 * half][n] = acc[t][r] * scale;
  }
  __syncthreads();

  // wave w owns rows 2w (lanes 0-15) and 2w+1 (lanes 16-31)
  const int row = 2 * wave + half;
  float mx = -1e30f;
  for (int j = lc; j < NC; j += 16) mx = fmaxf(mx, lgt[row][j]);
  for (int o = 8; o > 0; o >>= 1)   mx = fmaxf(mx, __shfl_xor(mx, o, 16));
  float sum = 0.f;
  for (int j = lc; j < NC; j += 16) {
    float e = __expf(lgt[row][j] - mx);
    lgt[row][j] = e;
    sum += e;
  }
  for (int o = 8; o > 0; o >>= 1) sum += __shfl_xor(sum, o, 16);
  const float inv = 1.0f / sum;
  _Float16* __restrict__ ar = att + ((size_t)b * NC + m0 + row) * NC;
  for (int j = lc; j < NC; j += 16) ar[j] = (_Float16)(lgt[row][j] * inv);
}

// ---- K5: ao = att @ v ; output transposed [b][d][i] -----------------------
// A = att[b][i][j] (ld C). B = vt[b][d][j] (ld C).
__global__ void av_gemm_kernel(const _Float16* __restrict__ att,
                               const _Float16* __restrict__ vt,
                               _Float16* __restrict__ aot) {
  const int b    = blockIdx.z;
  const int lane = threadIdx.x & 31;
  const int wave = threadIdx.x >> 5;
  const int m0   = blockIdx.y * 32  + (wave >> 2) * 16;   // i
  const int n0   = blockIdx.x * 256 + (wave & 3) * 64;    // d
  const _Float16* __restrict__ Am = att + (size_t)b * NC * NC;
  const _Float16* __restrict__ Bm = vt  + (size_t)b * NHW * NC;

  v8f acc[4] = {};
  for (int k0 = 0; k0 < NC; k0 += 32) {
    v16h af  = load_frag_a(Am, NC, m0, k0, lane);
    v16h bf0 = load_frag_b(Bm, NC, n0,      k0, lane);
    v16h bf1 = load_frag_b(Bm, NC, n0 + 16, k0, lane);
    v16h bf2 = load_frag_b(Bm, NC, n0 + 32, k0, lane);
    v16h bf3 = load_frag_b(Bm, NC, n0 + 48, k0, lane);
    acc[0] = wmma16(af, bf0, acc[0]);
    acc[1] = wmma16(af, bf1, acc[1]);
    acc[2] = wmma16(af, bf2, acc[2]);
    acc[3] = wmma16(af, bf3, acc[3]);
  }
  const int half = lane >> 4, lc = lane & 15;
  #pragma unroll
  for (int t = 0; t < 4; ++t) {
    const int n = n0 + 16 * t + lc;
    #pragma unroll
    for (int r = 0; r < 8; ++r) {
      const int m = m0 + r + 8 * half;
      aot[((size_t)b * NHW + n) * NC + m] = (_Float16)acc[t][r];
    }
  }
}

// ---- K6: out = w_out @ ao + b_out + x (residual), fp32 out ----------------
// A = w_out f16 [C,C] (ld C). B = aot [p][i] (ld C).
__global__ void outproj_kernel(const _Float16* __restrict__ wo,
                               const _Float16* __restrict__ aot,
                               const float* __restrict__ bias,
                               const float* __restrict__ x,
                               float* __restrict__ out) {
  const int b    = blockIdx.z;
  const int lane = threadIdx.x & 31;
  const int wave = threadIdx.x >> 5;
  const int m0   = blockIdx.y * 32  + (wave >> 2) * 16;   // o
  const int n0   = blockIdx.x * 256 + (wave & 3) * 64;    // p
  const _Float16* __restrict__ Bm = aot + (size_t)b * NHW * NC;

  v8f acc[4] = {};
  for (int k0 = 0; k0 < NC; k0 += 32) {
    v16h af  = load_frag_a(wo, NC, m0, k0, lane);
    v16h bf0 = load_frag_b(Bm, NC, n0,      k0, lane);
    v16h bf1 = load_frag_b(Bm, NC, n0 + 16, k0, lane);
    v16h bf2 = load_frag_b(Bm, NC, n0 + 32, k0, lane);
    v16h bf3 = load_frag_b(Bm, NC, n0 + 48, k0, lane);
    acc[0] = wmma16(af, bf0, acc[0]);
    acc[1] = wmma16(af, bf1, acc[1]);
    acc[2] = wmma16(af, bf2, acc[2]);
    acc[3] = wmma16(af, bf3, acc[3]);
  }
  const int half = lane >> 4, lc = lane & 15;
  #pragma unroll
  for (int t = 0; t < 4; ++t) {
    const int n = n0 + 16 * t + lc;
    #pragma unroll
    for (int r = 0; r < 8; ++r) {
      const int m = m0 + r + 8 * half;
      const size_t idx = ((size_t)b * NC + m) * NHW + n;
      out[idx] = acc[t][r] + bias[m] + x[idx];
    }
  }
}

// ---------------------------------------------------------------------------
extern "C" void kernel_launch(void* const* d_in, const int* in_sizes, int n_in,
                              void* d_out, int out_size, void* d_ws, size_t ws_size,
                              hipStream_t stream) {
  const float* x     = (const float*)d_in[0];
  const float* gamma = (const float*)d_in[1];
  const float* beta  = (const float*)d_in[2];
  const float* w_qkv = (const float*)d_in[3];
  const float* b_qkv = (const float*)d_in[4];
  const float* w_out = (const float*)d_in[5];
  const float* b_out = (const float*)d_in[6];
  float* out = (float*)d_out;

  // workspace layout (f16 elements)
  _Float16* ws = (_Float16*)d_ws;
  const size_t o_wqkv = 0;                                   // 3C*C
  const size_t o_wout = o_wqkv + (size_t)3 * NC * NC;        // C*C
  const size_t o_xnt  = o_wout + (size_t)NC * NC;            // B*HW*C
  const size_t o_q    = o_xnt  + (size_t)NB * NHW * NC;      // B*C*HW
  const size_t o_k    = o_q    + (size_t)NB * NC * NHW;      // B*C*HW
  const size_t o_vt   = o_k    + (size_t)NB * NC * NHW;      // B*HW*C
  const size_t o_att  = o_vt   + (size_t)NB * NHW * NC;      // B*C*C
  const size_t o_aot  = o_xnt;  // reuse xn_t region (dead after QKV GEMM)

  // K1: weight conversion  (3C*C + C*C = 2359296 elems = 9216 * 256)
  cvt_weights_kernel<<<dim3(9216), dim3(256), 0, stream>>>(
      w_qkv, w_out, ws + o_wqkv, ws + o_wout);

  // K2: GroupNorm -> xn_t f16
  groupnorm_kernel<<<dim3(NG, NB), dim3(256), 0, stream>>>(
      x, gamma, beta, ws + o_xnt);

  // K3: QKV GEMM  (M=2304 -> 72 m-blocks of 32; N=1024 -> 4 n-blocks of 256)
  qkv_gemm_kernel<<<dim3(4, 72, NB), dim3(256), 0, stream>>>(
      ws + o_wqkv, ws + o_xnt, b_qkv, ws + o_q, ws + o_k, ws + o_vt);

  // K4: fused Q.K^T * scale + softmax  (48 row panels per batch)
  att_softmax_kernel<<<dim3(NC / 16, NB), dim3(256), 0, stream>>>(
      ws + o_q, ws + o_k, ws + o_att);

  // K5: att @ v -> ao_t  (M=768 -> 24 m-blocks; N=1024 -> 4 n-blocks)
  av_gemm_kernel<<<dim3(4, 24, NB), dim3(256), 0, stream>>>(
      ws + o_att, ws + o_vt, ws + o_aot);

  // K6: out-proj + bias + residual
  outproj_kernel<<<dim3(4, 24, NB), dim3(256), 0, stream>>>(
      ws + o_wout, ws + o_aot, b_out, x, out);
}
